// ReferenceFFN_36009005809980
// MI455X (gfx1250) — matrix operand order repaired and checked by reading, not compiled, not run on hardware
//
#include <hip/hip_runtime.h>

// ---------------------------------------------------------------------------
// Sparse-FFN (top-k gated MLP) for MI455X / gfx1250, bf16 WMMA pipeline:
//   1) convert x -> bf16; convert+TRANSPOSE w_gate, w_up, w_down -> bf16
//      (K-major weights so all WMMA fragment loads are ds_load_b128)
//   2) build per-row activity bitmask from topk_indices
//   3) fused GEMM: G = x@w_gate, U = x@w_up -> Z = mask ? silu(G)*U : 0 (bf16)
//   4) GEMM: out = Z @ w_down (f32)
// GEMM tiles are double-buffered in LDS and filled with
// global_load_async_to_lds_b128 (ASYNCcnt), overlapping HBM with WMMA.
// ---------------------------------------------------------------------------

typedef __attribute__((ext_vector_type(16))) __bf16 v16bf;
typedef __attribute__((ext_vector_type(8)))  float  v8f;

static constexpr int MDIM   = 8192;
static constexpr int DMODEL = 2048;
static constexpr int DFFN   = 8192;
static constexpr int TOPK   = 512;
static constexpr int MASKW  = DFFN / 32;   // 256 mask words per row

static constexpr int LDA = 40;             // padded LDS stride (halfwords)
static constexpr int LDB = 40;             // 80B: 16B-aligned, conflict-free

union ABFrag {
    v16bf          v;
    uint4          q[2];
    unsigned int   u[8];
    unsigned short h[16];
};

__device__ __forceinline__ unsigned short f2bf(float f) {
    unsigned int u = __float_as_uint(f);
    unsigned int r = 0x7FFFu + ((u >> 16) & 1u);   // round-to-nearest-even
    return (unsigned short)((u + r) >> 16);
}

// 16B global -> LDS direct copy, tracked with ASYNCcnt (no VGPR staging).
__device__ __forceinline__ void async_b128(unsigned lds_off, const void* g) {
    asm volatile("global_load_async_to_lds_b128 %0, %1, off"
                 :: "v"(lds_off), "v"(g) : "memory");
}

// ------------------------------------------------------------ x conversion --
__global__ void cvt_f32_bf16(const float* __restrict__ s,
                             unsigned short* __restrict__ d, size_t n4) {
    size_t i = (size_t)blockIdx.x * blockDim.x + threadIdx.x;
    if (i >= n4) return;
    float4 f = ((const float4*)s)[i];
    ushort2 lo; lo.x = f2bf(f.x); lo.y = f2bf(f.y);
    ushort2 hi; hi.x = f2bf(f.z); hi.y = f2bf(f.w);
    ((ushort2*)d)[2 * i]     = lo;
    ((ushort2*)d)[2 * i + 1] = hi;
}

// ------------------------------------------- weight convert + transpose -----
// src: R x C f32 row-major ; dst: C x R bf16 row-major (i.e. src^T)
__global__ __launch_bounds__(256)
void cvt_transpose_bf16(const float* __restrict__ src,
                        unsigned short* __restrict__ dst, int R, int C) {
    __shared__ __align__(16) unsigned short t[64][72];
    const int r0 = blockIdx.y * 64;
    const int c0 = blockIdx.x * 64;
    const int tr = threadIdx.x >> 2;        // 0..63
    const int tq = (threadIdx.x & 3) * 16;  // 0,16,32,48

    const float* sp = src + (size_t)(r0 + tr) * C + c0 + tq;
#pragma unroll
    for (int i = 0; i < 16; i += 4) {
        float4 f = *(const float4*)(sp + i);
        t[tr][tq + i + 0] = f2bf(f.x);
        t[tr][tq + i + 1] = f2bf(f.y);
        t[tr][tq + i + 2] = f2bf(f.z);
        t[tr][tq + i + 3] = f2bf(f.w);
    }
    __syncthreads();

    ABFrag o;
#pragma unroll
    for (int i = 0; i < 16; ++i)
        o.h[i] = t[tq + i][tr];
    uint4* dp = (uint4*)(dst + (size_t)(c0 + tr) * R + r0 + tq);
    dp[0] = o.q[0];
    dp[1] = o.q[1];
}

// ------------------------------------------------------------------- mask ---
__global__ void zero_u32(unsigned int* __restrict__ p, size_t n) {
    size_t i = (size_t)blockIdx.x * blockDim.x + threadIdx.x;
    if (i < n) p[i] = 0u;
}

__global__ void build_mask(const int* __restrict__ idx,
                           unsigned int* __restrict__ mask) {
    size_t gid = (size_t)blockIdx.x * blockDim.x + threadIdx.x;
    if (gid >= (size_t)MDIM * TOPK) return;
    int m = (int)(gid >> 9);                 // TOPK == 512
    int j = idx[gid];                        // column in [0, DFFN)
    atomicOr(&mask[(size_t)m * MASKW + (j >> 5)], 1u << (j & 31));
}

// ---------------------------------------------------------------- GEMMs -----
// C[M,N] = A[M,K] @ B[K,N], with B supplied TRANSPOSED (BT[N,K], K-major).
// Block tile 128(M) x 64(N), K-step 32, 256 threads = 8 waves (wave32),
// wave grid 4x2, each wave: 2x2 subtiles of v_wmma_f32_16x16x32_bf16.
// Double-buffered LDS, filled by global_load_async_to_lds_b128.
// FUSED=1: second BT (w_up) shares A fragments; masked-SiLU epilogue -> bf16.
template <int FUSED>
__global__ __launch_bounds__(256)
void gemm_wmma(const unsigned short* __restrict__ A,
               const unsigned short* __restrict__ BT0,
               const unsigned short* __restrict__ BT1,
               const unsigned int* __restrict__ mask,
               unsigned short* __restrict__ Zout,
               float* __restrict__ Fout,
               int N, int K) {
    __shared__ __align__(16) unsigned short As[2][128 * LDA];      // 20 KB
    __shared__ __align__(16) unsigned short Bs[2][2 * 64 * LDB];   // 20 KB

    const int tid  = threadIdx.x;
    const int lane = tid & 31;
    const int wave = tid >> 5;
    const int lr   = lane & 15;
    const int hi   = lane >> 4;
    const int wrow = (wave & 3) * 32;
    const int wcol = (wave >> 2) * 32;
    const int m0   = blockIdx.y * 128;
    const int n0   = blockIdx.x * 64;

    v8f acc0[2][2] = {};
    v8f acc1[2][2] = {};

    const int aRow  = tid >> 1;
    const int aHalf = (tid & 1) * 16;
    const int akb   = hi * 16;              // A per-lane K base (contiguous k = 16*hi + s)
    const int bkb   = hi * 16;              // B per-lane K base (contiguous k = 16*hi + s)

    // Issue one double-buffer tile (per-wave ASYNCcnt += FUSED ? 4 : 3).
    auto issue_tile = [&](int buf, int k0) {
        const unsigned short* ag = A + (size_t)(m0 + aRow) * K + k0 + aHalf;
        unsigned ad = (unsigned)(size_t)&As[buf][aRow * LDA + aHalf];
        async_b128(ad,      ag);
        async_b128(ad + 16, ag + 8);
        if (FUSED) {
            const int mat = tid >> 7;            // uniform per wave
            const int t   = tid & 127;
            const int br  = t >> 1;
            const int bh  = (t & 1) * 16;
            const unsigned short* bg =
                (mat ? BT1 : BT0) + (size_t)(n0 + br) * K + k0 + bh;
            unsigned bd =
                (unsigned)(size_t)&Bs[buf][mat * 64 * LDB + br * LDB + bh];
            async_b128(bd,      bg);
            async_b128(bd + 16, bg + 8);
        } else {
            const int br = tid >> 2;
            const int bq = (tid & 3) * 8;
            const unsigned short* bg = BT0 + (size_t)(n0 + br) * K + k0 + bq;
            unsigned bd = (unsigned)(size_t)&Bs[buf][br * LDB + bq];
            async_b128(bd, bg);
        }
    };

    issue_tile(0, 0);
    int cur = 0;
    for (int k0 = 0; k0 < K; k0 += 32) {
        const bool has_next = (k0 + 32 < K);
        if (has_next)
            issue_tile(cur ^ 1, k0 + 32);   // prev barrier freed this buffer

        // Async loads retire in order: <= (next tile's count) outstanding
        // means the current tile is fully in LDS.
        if (has_next) {
            if (FUSED) asm volatile("s_wait_asynccnt 0x4" ::: "memory");
            else       asm volatile("s_wait_asynccnt 0x3" ::: "memory");
        } else {
            asm volatile("s_wait_asynccnt 0x0" ::: "memory");
        }
        __syncthreads();

        // ---- A fragments: two contiguous 8-elem runs -> 2x ds_load_b128
        ABFrag afrag[2];
#pragma unroll
        for (int i = 0; i < 2; ++i) {
            const unsigned short* ap =
                &As[cur][(wrow + i * 16 + lr) * LDA + akb];
            afrag[i].q[0] = ((const uint4*)ap)[0];        // K = akb..akb+7
            afrag[i].q[1] = *(const uint4*)(ap + 8);      // K = akb+8..+15
        }
        // ---- B fragments: 16 K-contiguous elements -> 2x ds_load_b128
#pragma unroll
        for (int j = 0; j < 2; ++j) {
            const int col = wcol + j * 16 + lr;
            ABFrag b0, b1;
            {
                const unsigned short* bp = &Bs[cur][col * LDB + bkb];
                b0.q[0] = ((const uint4*)bp)[0];
                b0.q[1] = ((const uint4*)bp)[1];
            }
            if (FUSED) {
                const unsigned short* bp =
                    &Bs[cur][64 * LDB + col * LDB + bkb];
                b1.q[0] = ((const uint4*)bp)[0];
                b1.q[1] = ((const uint4*)bp)[1];
            }
#pragma unroll
            for (int i = 0; i < 2; ++i) {
                acc0[i][j] = __builtin_amdgcn_wmma_f32_16x16x32_bf16(
                    false, afrag[i].v, false, b0.v, (short)0, acc0[i][j],
                    false, false);
                if (FUSED)
                    acc1[i][j] = __builtin_amdgcn_wmma_f32_16x16x32_bf16(
                        false, afrag[i].v, false, b1.v, (short)0, acc1[i][j],
                        false, false);
            }
        }
        __syncthreads();   // all waves done reading `cur` before it refills
        cur ^= 1;
    }

    // ---- epilogue: C/D layout -> row = base + vgpr + 8*hi, col = base + lr
#pragma unroll
    for (int i = 0; i < 2; ++i) {
#pragma unroll
        for (int j = 0; j < 2; ++j) {
#pragma unroll
            for (int v = 0; v < 8; ++v) {
                const int r = m0 + wrow + i * 16 + v + hi * 8;
                const int c = n0 + wcol + j * 16 + lr;
                if (FUSED) {
                    float g = acc0[i][j][v];
                    float u = acc1[i][j][v];
                    unsigned int mw = mask[(size_t)r * MASKW + (c >> 5)];
                    float z = ((mw >> (c & 31)) & 1u)
                                  ? g * u / (1.0f + __expf(-g))
                                  : 0.0f;
                    Zout[(size_t)r * DFFN + c] = f2bf(z);
                } else {
                    Fout[(size_t)r * N + c] = acc0[i][j][v];
                }
            }
        }
    }
}

// ---------------------------------------------------------------------------
extern "C" void kernel_launch(void* const* d_in, const int* in_sizes, int n_in,
                              void* d_out, int out_size, void* d_ws,
                              size_t ws_size, hipStream_t stream) {
    const float* x  = (const float*)d_in[0];
    const int*   ix = (const int*)d_in[1];
    const float* wg = (const float*)d_in[2];
    const float* wu = (const float*)d_in[3];
    const float* wd = (const float*)d_in[4];
    float*       out = (float*)d_out;

    // workspace layout (bf16 copies + bitmask + Z)
    unsigned short* xb   = (unsigned short*)d_ws;                 // [M][DMODEL]
    unsigned short* wgbT = xb   + (size_t)MDIM * DMODEL;          // [DFFN][DMODEL]
    unsigned short* wubT = wgbT + (size_t)DMODEL * DFFN;          // [DFFN][DMODEL]
    unsigned short* wdbT = wubT + (size_t)DMODEL * DFFN;          // [DMODEL][DFFN]
    unsigned int*   msk  = (unsigned int*)(wdbT + (size_t)DFFN * DMODEL);
    unsigned short* Zb   = (unsigned short*)(msk + (size_t)MDIM * MASKW);

    const size_t nx = (size_t)MDIM * DMODEL;

    cvt_f32_bf16<<<(unsigned)((nx / 4 + 255) / 256), 256, 0, stream>>>(
        x, xb, nx / 4);

    // w_gate/w_up: [DMODEL][DFFN] -> transposed bf16 [DFFN][DMODEL]
    dim3 gt1(DFFN / 64, DMODEL / 64);
    cvt_transpose_bf16<<<gt1, 256, 0, stream>>>(wg, wgbT, DMODEL, DFFN);
    cvt_transpose_bf16<<<gt1, 256, 0, stream>>>(wu, wubT, DMODEL, DFFN);
    // w_down: [DFFN][DMODEL] -> transposed bf16 [DMODEL][DFFN]
    dim3 gt2(DMODEL / 64, DFFN / 64);
    cvt_transpose_bf16<<<gt2, 256, 0, stream>>>(wd, wdbT, DFFN, DMODEL);

    const size_t nm = (size_t)MDIM * MASKW;
    zero_u32<<<(unsigned)((nm + 255) / 256), 256, 0, stream>>>(msk, nm);
    build_mask<<<(unsigned)(((size_t)MDIM * TOPK + 255) / 256), 256, 0,
                 stream>>>(ix, msk);

    dim3 g1(DFFN / 64, MDIM / 128);   // fused gate/up -> Z
    gemm_wmma<1><<<g1, 256, 0, stream>>>(xb, wgbT, wubT, msk, Zb, nullptr,
                                         DFFN, DMODEL);

    dim3 g2(DMODEL / 64, MDIM / 128); // Z @ w_down -> out
    gemm_wmma<0><<<g2, 256, 0, stream>>>(Zb, wdbT, nullptr, nullptr, nullptr,
                                         out, DMODEL, DFFN);
}
